// CustomLSTM_31894427140504
// MI455X (gfx1250) — matrix operand order repaired and checked by reading, not compiled, not run on hardware
//
#include <hip/hip_runtime.h>
#include <hip/hip_bf16.h>

typedef _Float16 v16h __attribute__((ext_vector_type(16)));
typedef float    v8f  __attribute__((ext_vector_type(8)));

#define B_ 2048
#define T_ 256
#define I_ 5
#define H_ 50
#define NT 13              // N tiles of 16 covering 4H=200 (col = 4*j + gate), padded to 208
#define WAVES NT           // one tile per wave
#define THREADS (WAVES*32) // 416

union V16U { uint4 q[2]; v16h v; };

__device__ __forceinline__ void lds_fence() {
  asm volatile("s_wait_dscnt 0" ::: "memory");
}

#if __has_builtin(__builtin_amdgcn_tanhf)
__device__ __forceinline__ float tanh_fast(float x) { return __builtin_amdgcn_tanhf(x); }
__device__ __forceinline__ float sigm(float x) {
  return fmaf(0.5f, tanh_fast(0.5f * x), 0.5f);   // v_mul + v_tanh + v_fma
}
#else
__device__ __forceinline__ float fast_rcp(float x) { return __builtin_amdgcn_rcpf(x); }
__device__ __forceinline__ float tanh_fast(float x) {
  return 1.0f - 2.0f * fast_rcp(__expf(2.0f * x) + 1.0f);
}
__device__ __forceinline__ float sigm(float x) {
  return fast_rcp(1.0f + __expf(-x));
}
#endif

__global__ __launch_bounds__(THREADS)
void lstm_scan_kernel(const float* __restrict__ x,
                      const float* Wxf, const float* bxf, const float* Wxi, const float* bxi,
                      const float* Wxc, const float* bxc, const float* Wxo, const float* bxo,
                      const float* Whf, const float* bhf, const float* Whi, const float* bhi,
                      const float* Whc, const float* bhc, const float* Who, const float* bho,
                      const float* bf, const float* bi, const float* bc, const float* bo,
                      const float* Wfc, const float* bfc,
                      float* __restrict__ out)
{
  // A operand rows [h(50) | x(5) | 1 | pad(8)], double buffered by step parity.
  __shared__ alignas(16) _Float16 hb[2][16][72];     // 4.6 KB
  __shared__ alignas(16) float gtile[NT][16][16];    // 13.3 KB per-wave C staging

  const int tid  = threadIdx.x;
  const int lane = tid & 31;
  const int w    = tid >> 5;                 // tile index 0..12
  const int R    = blockIdx.x * 16;          // this block's 16 batch rows
  const int m16  = lane & 15;
  const int hi16 = (lane & 16) ? 1 : 0;

  // ---- build this wave's B fragments directly in registers (contiguous K: lane half h holds K 16h..16h+15) ----
  // Fused weight Wt[k][col], col = 4*j + gate:
  //   k<50: Wh_g[j][k]; 50<=k<55: Wx_g[j][k-50]; k==55: bx_g[j]+bh_g[j]+b_g[j]; else 0.
  V16U b0, b1;
  {
    const int col = w * 16 + m16;
    const int j = col >> 2, g = col & 3;
    const float* whp = (g==0)?Whf:(g==1)?Whi:(g==2)?Whc:Who;
    const float* wxp = (g==0)?Wxf:(g==1)?Wxi:(g==2)?Wxc:Wxo;
    const float* bhp = (g==0)?bhf:(g==1)?bhi:(g==2)?bhc:bho;
    const float* bxp = (g==0)?bxf:(g==1)?bxi:(g==2)?bxc:bxo;
    const float* bgp = (g==0)?bf :(g==1)?bi :(g==2)?bc :bo;
    unsigned int ub[2][8];
    #pragma unroll
    for (int ks = 0; ks < 2; ++ks) {
      #pragma unroll
      for (int dw = 0; dw < 8; ++dw) {
        int kb = ks*32 + hi16*16 + dw*2;     // K of low half; +1 for high half
        float f[2] = {0.f, 0.f};
        if (j < H_) {
          #pragma unroll
          for (int s = 0; s < 2; ++s) {
            int k = kb + s;
            float v = 0.f;
            if (k < H_)            v = whp[j*H_ + k];
            else if (k < H_ + I_)  v = wxp[j*I_ + (k - H_)];
            else if (k == H_ + I_) v = bxp[j] + bhp[j] + bgp[j];
            f[s] = v;
          }
        }
        _Float16 h2[2] = {(_Float16)f[0], (_Float16)f[1]};
        __builtin_memcpy(&ub[ks][dw], h2, 4);
      }
    }
    __builtin_memcpy(&b0, &ub[0][0], 32);
    __builtin_memcpy(&b1, &ub[1][0], 32);
  }

  // ---- init both A buffers: h0 = 0, bias lane (col 55) = 1 ----
  for (int idx = tid; idx < 2*16*72; idx += THREADS) {
    int k = idx % 72;
    ((_Float16*)hb)[idx] = (k == (H_ + I_)) ? (_Float16)1.0f : (_Float16)0.0f;
  }

  float xv[I_] = {0.f, 0.f, 0.f, 0.f, 0.f};
  if (w == 0 && lane < 16) {
    const float* xr = x + (long)(R + lane) * T_ * I_;
    #pragma unroll
    for (int i = 0; i < I_; ++i) hb[0][lane][H_ + i] = (_Float16)xr[i];   // x(0)
    #pragma unroll
    for (int i = 0; i < I_; ++i) xv[i] = xr[I_ + i];                       // prefetch x(1)
  }
  __syncthreads();

  float creg[2] = {0.f, 0.f};
  const int kbA    = hi16;          // uint4 offset: lane half h reads K 16h..16h+15 (contiguous)
  const int jjbase = lane >> 4;     // 0 or 1
  const int r0     = hi16 * 8;

  // Unconditional h-store columns: j >= 50 (tile 12, jj=2,3) remaps to dead
  // padding columns 56/57, whose B rows are zero — never contributes to the GEMM.
  const int j0  = 4*w + jjbase;
  const int j1  = j0 + 2;
  const int js0 = (j0 < H_) ? j0 : (j0 + 6);
  const int js1 = (j1 < H_) ? j1 : (j1 + 6);

  for (int t = 0; t < T_; ++t) {
    const int cur = t & 1, nxt = cur ^ 1;

    // Load A fragments (K=0..31, K=32..63) for [h(t) | x(t) | 1]
    const uint4* row = (const uint4*)(&hb[cur][m16][0]);
    V16U a0, a1;
    a0.q[0] = row[0 + 2*kbA];
    a0.q[1] = row[1 + 2*kbA];
    a1.q[0] = row[4 + 2*kbA];
    a1.q[1] = row[5 + 2*kbA];

    v8f acc = {};
    acc = __builtin_amdgcn_wmma_f32_16x16x32_f16(false, a0.v, false, b0.v,
                                                 (short)0, acc, false, false);
    acc = __builtin_amdgcn_wmma_f32_16x16x32_f16(false, a1.v, false, b1.v,
                                                 (short)0, acc, false, false);

    #pragma unroll
    for (int r = 0; r < 8; ++r) gtile[w][r0 + r][m16] = acc[r];
    lds_fence();                    // RAW: C visible cross-lane (same wave, DS in-order)

    // gate update: tile w holds complete f,i,c,o quads for j = 4w..4w+3
    {
      const float4 gq0 = *(const float4*)(&gtile[w][m16][jjbase*4]);
      float cn0 = sigm(gq0.x) * creg[0] + sigm(gq0.y) * tanh_fast(gq0.z);
      creg[0] = cn0;
      hb[nxt][m16][js0] = (_Float16)(sigm(gq0.w) * tanh_fast(cn0));   // ds_store_b16

      const float4 gq1 = *(const float4*)(&gtile[w][m16][(jjbase + 2)*4]);
      float cn1 = sigm(gq1.x) * creg[1] + sigm(gq1.y) * tanh_fast(gq1.z);
      creg[1] = cn1;
      hb[nxt][m16][js1] = (_Float16)(sigm(gq1.w) * tanh_fast(cn1));   // ds_store_b16
    }

    // stream x into the next buffer: store x(t+1), prefetch x(t+2)
    if (w == 0 && lane < 16) {
      if (t + 1 < T_) {
        #pragma unroll
        for (int i = 0; i < I_; ++i) hb[nxt][lane][H_ + i] = (_Float16)xv[i];
      }
      if (t + 2 < T_) {
        const float* xr = x + ((long)(R + lane) * T_ + (t + 2)) * I_;
        #pragma unroll
        for (int i = 0; i < I_; ++i) xv[i] = xr[i];
      }
    }
    __syncthreads();                // hb[nxt] complete before next step reads it
  }

  // ---- final projection: out[b] = h_T . Wfc + bfc ----
  // Last step (t = T_-1) wrote h into buffer (T_ & 1) == 0.
  if (w == 0 && lane < 16) {
    float s = bfc[0];
    for (int j = 0; j < H_; ++j) s += (float)hb[T_ & 1][lane][j] * Wfc[j];
    out[R + lane] = s;
  }
}

extern "C" void kernel_launch(void* const* d_in, const int* in_sizes, int n_in,
                              void* d_out, int out_size, void* d_ws, size_t ws_size,
                              hipStream_t stream) {
  (void)in_sizes; (void)n_in; (void)d_ws; (void)ws_size; (void)out_size;
  const float* p[23];
  for (int i = 0; i < 23; ++i) p[i] = (const float*)d_in[i];

  dim3 grid(B_ / 16), block(THREADS);
  lstm_scan_kernel<<<grid, block, 0, stream>>>(
      p[0],
      p[1],  p[2],  p[3],  p[4],  p[5],  p[6],  p[7],  p[8],
      p[9],  p[10], p[11], p[12], p[13], p[14], p[15], p[16],
      p[17], p[18], p[19], p[20],
      p[21], p[22],
      (float*)d_out);
}
